// Gesture_87170656239809
// MI455X (gfx1250) — matrix operand order repaired and checked by reading, not compiled, not run on hardware
//
#include <hip/hip_runtime.h>

typedef _Float16 v16h __attribute__((ext_vector_type(16)));
typedef _Float16 v8h  __attribute__((ext_vector_type(8)));
typedef _Float16 h2   __attribute__((ext_vector_type(2)));
typedef __fp16   f16x2 __attribute__((ext_vector_type(2)));
typedef float    v8f  __attribute__((ext_vector_type(8)));

#define NW 20            // 6 L1 + 6 L2 + 3 merged L3 + 5 head
#define WAVES_PER_BLK 8

struct KArgs {
  const float* x;        // [B,63]
  float* out;            // [B,10]
  const float* p[42];    // 7 modules x (w1,b1,w2,b2,w3,b3)
  int ntiles;            // B/16
};

__device__ __forceinline__ v8f bcast8(float v) {
  v8f r;
#pragma unroll
  for (int i = 0; i < 8; ++i) r[i] = v;
  return r;
}

__device__ __forceinline__ h2 h2zero() {
  h2 z; z[0] = (_Float16)0.0f; z[1] = (_Float16)0.0f; return z;
}

// v_cvt_pk_rtz_f16_f32: two f32 -> packed f16 pair
__device__ __forceinline__ h2 cvt2(float a, float b) {
  f16x2 p = __builtin_amdgcn_cvt_pkrtz(a, b);
  return __builtin_bit_cast(h2, p);
}

__device__ __forceinline__ v8f wmma16(v16h a, v16h b, v8f c) {
  return __builtin_amdgcn_wmma_f32_16x16x32_f16(
      false, a, false, b, (short)0, c, false, false);
}

// cvt_pk_rtz + v_pk_max relu + b16/d16_hi stores; `keep`=false stores zeros.
__device__ __forceinline__ void store_relu_col(_Float16* t, v8f a, bool keep,
                                               int lane, int col) {
  int rb = (lane < 16) ? 0 : 8;
  h2 z = h2zero();
#pragma unroll
  for (int j = 0; j < 4; ++j) {
    h2 ph = cvt2(a[2 * j], a[2 * j + 1]);
    ph = __builtin_elementwise_max(ph, z);   // v_pk_max_num_f16
    ph = keep ? ph : z;
    t[(rb + 2 * j) * 32 + col] = ph[0];
    t[(rb + 2 * j + 1) * 32 + col] = ph[1];
  }
}

// A-fragment, K0-31 from tile columns [coff, coff+32). Reads past valid columns
// are fine: tile memory is always finite f16 and the matching B rows are zero.
__device__ __forceinline__ v16h loadA_full(const _Float16* t, int lane, int coff) {
  int m = lane & 15;
  int o = (lane < 16) ? 0 : 8;
  v8h lo8 = *(const v8h*)&t[m * 32 + coff + o];
  v8h hi8 = *(const v8h*)&t[m * 32 + coff + 16 + o];
  v16h r;
#pragma unroll
  for (int i = 0; i < 8; ++i) { r[i] = lo8[i]; r[8 + i] = hi8[i]; }
  return r;
}

// Per-tile fragment reload from LDS; `fo` is laundered so this cannot be hoisted.
__device__ __forceinline__ v16h loadFrag(const char* wbase, int w, int fo) {
  const v8h* q = (const v8h*)(wbase + w * 1024 + fo);
  v8h lo8 = q[0], hi8 = q[1];
  v16h r;
#pragma unroll
  for (int i = 0; i < 8; ++i) { r[i] = lo8[i]; r[8 + i] = hi8[i]; }
  return r;
}

// Assemble A fragment low half from packed pairs (upper K half zero)
__device__ __forceinline__ v16h packA8(h2 p0, h2 p1, h2 p2, h2 p3) {
  v16h a;
  a[0] = p0[0]; a[1] = p0[1]; a[2] = p1[0]; a[3] = p1[1];
  a[4] = p2[0]; a[5] = p2[1]; a[6] = p3[0]; a[7] = p3[1];
#pragma unroll
  for (int h = 8; h < 16; ++h) a[h] = (_Float16)0.0f;
  return a;
}

__global__ __launch_bounds__(256, 4)
void gesture_mlp_kernel(KArgs A) {
  __shared__ __align__(16) _Float16 s_wf[NW][32][16];  // packed B fragments
  __shared__ float s_wb[NW][32];                       // per-lane biases
  __shared__ __align__(16) _Float16 s_act[WAVES_PER_BLK][512];
  __shared__ __align__(16) _Float16 s_hin[WAVES_PER_BLK][512];

  // palm K -> feature-index permutation (K chosen so gather uses shared voffsets:
  // K0-5 cols{3,4,5,15,16,17} (+36 for hi lanes -> K8-13 cols{39,40,41,51,52,53}),
  // K6,7 cols{0,1} (+28 -> K14,15 cols{28,29}), K16,17 cols{2,27})
  const int PKJ[18] = {3, 4, 5, 6, 7, 8,  0, 1,
                       12, 13, 14, 15, 16, 17,  10, 11,  2, 9};

  // Zero activation/concat tiles so any over-read is finite (see loadA_full).
  {
    unsigned* za = (unsigned*)&s_act[0][0];
    unsigned* zh = (unsigned*)&s_hin[0][0];
    for (int i = threadIdx.x; i < WAVES_PER_BLK * 256; i += blockDim.x) {
      za[i] = 0u; zh[i] = 0u;
    }
  }

  // ---- one-time cooperative weight packing (B layout: lane n=l%16 is column,
  //      lanes 0-15 hold K=0..7,16..23, lanes 16-31 hold K=8..15,24..31) ----
  for (int idx = threadIdx.x; idx < NW * 32; idx += blockDim.x) {
    int w = idx >> 5, l = idx & 31;
    int n = l & 15, kb = (l < 16) ? 0 : 8;
    float frag[16];
    float bias = 0.0f;
    if (w == 0) {                      // palm L1 (uses K permutation)
      const float* wp = A.p[0];
      const float* bp = A.p[1];
#pragma unroll
      for (int h = 0; h < 16; ++h) {
        int k = kb + h + (h >= 8 ? 8 : 0);
        frag[h] = (n < 15 && k < 18) ? wp[n * 18 + PKJ[k]] : 0.0f;
      }
      bias = (n < 15) ? bp[n] : 0.0f;
    } else if (w < 6) {                // finger L1
      const float* wp = A.p[w * 6 + 0];
      const float* bp = A.p[w * 6 + 1];
#pragma unroll
      for (int h = 0; h < 16; ++h) { int k = kb + h + (h >= 8 ? 8 : 0); frag[h] = (n < 15 && k < 12) ? wp[n * 12 + k] : 0.0f; }
      bias = (n < 15) ? bp[n] : 0.0f;
    } else if (w < 12) {               // branch L2 (15->10)
      int br = w - 6;
      const float* wp = A.p[br * 6 + 2];
      const float* bp = A.p[br * 6 + 3];
#pragma unroll
      for (int h = 0; h < 16; ++h) { int k = kb + h + (h >= 8 ? 8 : 0); frag[h] = (n < 10 && k < 15) ? wp[n * 15 + k] : 0.0f; }
      bias = (n < 10) ? bp[n] : 0.0f;
    } else if (w < 15) {               // merged L3 pair p: brB at K0-9, brA at K16-25
      int p = w - 12, ba = 2 * p, bb = 2 * p + 1;
      const float* wa = A.p[ba * 6 + 4]; const float* bpa = A.p[ba * 6 + 5];
      const float* wb = A.p[bb * 6 + 4]; const float* bpb = A.p[bb * 6 + 5];
#pragma unroll
      for (int h = 0; h < 16; ++h) {
        int k = kb + h + (h >= 8 ? 8 : 0); float v = 0.0f;
        if (n < 5 && k >= 16 && k < 26) v = wa[n * 10 + (k - 16)];
        if (n >= 5 && n < 10 && k < 10) v = wb[(n - 5) * 10 + k];
        frag[h] = v;
      }
      bias = (n < 5) ? bpa[n] : ((n < 10) ? bpb[n - 5] : 0.0f);
    } else {                           // head: 15,16=L1(n0-15,n16-19) 17,18=L2 19=L3
      int lay = (w < 17) ? 0 : ((w < 19) ? 1 : 2);
      int noff = (w == 16 || w == 18) ? 16 : 0;
      int din = (lay == 0) ? 30 : 20;
      int dout = (lay == 2) ? 10 : 20;
      const float* wp = A.p[36 + lay * 2];
      const float* bp = A.p[36 + lay * 2 + 1];
      int nn = noff + n;
#pragma unroll
      for (int h = 0; h < 16; ++h) { int k = kb + h + (h >= 8 ? 8 : 0); frag[h] = (nn < dout && k < din) ? wp[nn * din + k] : 0.0f; }
      bias = (nn < dout) ? bp[nn] : 0.0f;
    }
#pragma unroll
    for (int h = 0; h < 16; ++h) s_wf[w][l][h] = (_Float16)frag[h];
    s_wb[w][l] = bias;
  }
  __syncthreads();

  const int lane = threadIdx.x & 31;
  const int wib = threadIdx.x >> 5;
  const int m = lane & 15;
  const bool lo = (lane < 16);
  const int gw = blockIdx.x * (blockDim.x >> 5) + wib;
  const int nwv = gridDim.x * (blockDim.x >> 5);

  float bs[NW];
#pragma unroll
  for (int w = 0; w < NW; ++w) bs[w] = s_wb[w][lane];

  _Float16* act = s_act[wib];
  _Float16* hin = s_hin[wib];

  const int kbe = lo ? 0 : 8;    // finger K base per lane half (elements)
  const int sub4 = lo ? 0 : 4;   // backoff keeps invalid lanes in-bounds
  const int d36 = lo ? 0 : 36;   // palm voffset deltas (K permutation translates)
  const int d28 = lo ? 0 : 28;

  const float* xp = A.x;
  const char* wbase = (const char*)&s_wf[0][0][0];

  for (int tile = gw; tile < A.ntiles; tile += nwv) {
    int rowel = (tile * 16 + m) * 63;
    int fA = rowel + kbe;
    int fB = fA - sub4;     // hi lanes read in-bounds (finite) x for zero B rows
    int vP = rowel + d36;
    int vQ = rowel + d28;
    int fo = lane * 32;
    asm volatile("" : "+v"(fo));  // defeat hoisting of per-tile fragment reloads

#pragma unroll
    for (int p = 0; p < 3; ++p) {
      const int ba = 2 * p, bb = 2 * p + 1;
      // ---- both L1 gathers (immediate-offset loads, packed conversions;
      //      no masking needed: out-of-range K hits zero rows of B) ----
      v16h aA, aB;
      if (ba == 0) {
        float t[10];
        t[0] = xp[vP + 3];  t[1] = xp[vP + 4];       // K0,1 / K8,9
        t[2] = xp[vP + 5];  t[3] = xp[vP + 15];      // K2,3 / K10,11
        t[4] = xp[vP + 16]; t[5] = xp[vP + 17];      // K4,5 / K12,13
        t[6] = xp[vQ + 0];  t[7] = xp[vQ + 1];       // K6,7 / K14,15
        t[8] = xp[rowel + 2]; t[9] = xp[rowel + 27]; // K16,17 (B rows 24,25 = 0)
        aA = packA8(cvt2(t[0], t[1]), cvt2(t[2], t[3]),
                    cvt2(t[4], t[5]), cvt2(t[6], t[7]));
        h2 p4 = cvt2(t[8], t[9]);
        aA[8] = p4[0]; aA[9] = p4[1];
      } else {
        const int base0 = 3 + 12 * (ba - 1);
        float t[8];
#pragma unroll
        for (int h = 0; h < 4; ++h) t[h] = xp[fA + base0 + h];
#pragma unroll
        for (int h = 4; h < 8; ++h) t[h] = xp[fB + base0 + h];
        aA = packA8(cvt2(t[0], t[1]), cvt2(t[2], t[3]),
                    cvt2(t[4], t[5]), cvt2(t[6], t[7]));
      }
      {
        const int base1 = 3 + 12 * (bb - 1);
        float u[8];
#pragma unroll
        for (int h = 0; h < 4; ++h) u[h] = xp[fA + base1 + h];
#pragma unroll
        for (int h = 4; h < 8; ++h) u[h] = xp[fB + base1 + h];
        aB = packA8(cvt2(u[0], u[1]), cvt2(u[2], u[3]),
                    cvt2(u[4], u[5]), cvt2(u[6], u[7]));
      }
      // ---- both L1 WMMAs (independent chains fill hazard slots) ----
      v8f c1A = wmma16(aA, loadFrag(wbase, ba, fo), bcast8(bs[ba]));
      v8f c1B = wmma16(aB, loadFrag(wbase, bb, fo), bcast8(bs[bb]));
      store_relu_col(act, c1A, true, lane, m);        // brA L1 -> cols 0-15
      store_relu_col(act, c1B, true, lane, 16 + m);   // brB L1 -> cols 16-31
      // ---- both L2 (full-width A; upper K columns hit zero B rows) ----
      v16h a2A = loadA_full(act, lane, 0);
      v16h a2B = loadA_full(act, lane, 16);
      v16h w2A = loadFrag(wbase, 6 + ba, fo);
      v16h w2B = loadFrag(wbase, 6 + bb, fo);
      v8f c2A = wmma16(a2A, w2A, bcast8(bs[6 + ba]));
      v8f c2B = wmma16(a2B, w2B, bcast8(bs[6 + bb]));
      store_relu_col(act, c2B, true, lane, m);        // brB L2 -> cols 0-15 (K0-9)
      store_relu_col(act, c2A, true, lane, 16 + m);   // brA L2 -> cols 16-31 (K16-25)
      // ---- merged L3 (no relu) ----
      v16h a3 = loadA_full(act, lane, 0);
      v8f c3 = wmma16(a3, loadFrag(wbase, 12 + p, fo), bcast8(bs[12 + p]));
      if (m < 10) {
        int rb = lo ? 0 : 8;
#pragma unroll
        for (int j = 0; j < 4; ++j) {
          h2 ph = cvt2(c3[2 * j], c3[2 * j + 1]);
          hin[(rb + 2 * j) * 32 + 10 * p + m] = ph[0];
          hin[(rb + 2 * j + 1) * 32 + 10 * p + m] = ph[1];
        }
      }
    }

    // ---- head: 30 -> 20 -> 20 -> 10 ----
    v16h ha = loadA_full(hin, lane, 0);
    v8f h0 = wmma16(ha, loadFrag(wbase, 15, fo), bcast8(bs[15]));
    v8f h1 = wmma16(ha, loadFrag(wbase, 16, fo), bcast8(bs[16]));
    store_relu_col(act, h0, true, lane, m);
    store_relu_col(act, h1, m < 4, lane, 16 + m);     // cols 20..31 -> 0
    v16h hb = loadA_full(act, lane, 0);
    v8f g0 = wmma16(hb, loadFrag(wbase, 17, fo), bcast8(bs[17]));
    v8f g1 = wmma16(hb, loadFrag(wbase, 18, fo), bcast8(bs[18]));
    store_relu_col(act, g0, true, lane, m);
    store_relu_col(act, g1, m < 4, lane, 16 + m);
    v16h hc = loadA_full(act, lane, 0);
    v8f o = wmma16(hc, loadFrag(wbase, 19, fo), bcast8(bs[19]));

    if (m < 10) {
      float* op = A.out + (size_t)tile * 160;
      int rb = lo ? 0 : 8;
#pragma unroll
      for (int v = 0; v < 8; ++v)
        op[(rb + v) * 10 + m] = o[v];
    }
  }
}

extern "C" void kernel_launch(void* const* d_in, const int* in_sizes, int n_in,
                              void* d_out, int out_size, void* d_ws, size_t ws_size,
                              hipStream_t stream) {
  (void)n_in; (void)out_size; (void)d_ws; (void)ws_size;
  KArgs a;
  a.x = (const float*)d_in[0];
  a.out = (float*)d_out;
  for (int i = 0; i < 42; ++i) a.p[i] = (const float*)d_in[1 + i];
  long B = (long)in_sizes[0] / 63;
  a.ntiles = (int)(B / 16);
  hipLaunchKernelGGL(gesture_mlp_kernel, dim3(2048), dim3(256), 0, stream, a);
}
